// Segment_Model_8942121911180
// MI455X (gfx1250) — compile-verified, matrix operand order checked by reading, not run on hardware
//
#include <hip/hip_runtime.h>
#include <hip/hip_bf16.h>

// ---------------------------------------------------------------------------
// CDNA5 (gfx1250) implementation. All matrix work goes through
// v_wmma_f32_16x16x32_f16 (f16 inputs, f32 accumulate). B operands are
// always stored N x Kp row-major so A and B fragments use the identical
// per-lane layout and load as 128-bit vectors. Sequential RNN recurrences
// are persistent single-workgroup scans with h in LDS.
// ---------------------------------------------------------------------------

typedef __attribute__((ext_vector_type(16))) _Float16     v16h;
typedef __attribute__((ext_vector_type(8)))  float        v8f;
typedef __attribute__((ext_vector_type(4)))  unsigned int v4u;

#define ACT_NONE 0
#define ACT_RELU 1
#define ACT_SELU 2

union HFrag { v16h v; unsigned int u[8]; v4u q[2]; };

// WMMA GEMM: C(MxN) = act(A(MxKp) * B'(N x Kp) + bias)
// Each wave computes a 16x32 output tile (two accumulators, shared A frag).
// Optional accumulate into f32 C with scale; optional f16 output copy.
__global__ __launch_bounds__(128) void wmma_gemm_kernel(
    const _Float16* __restrict__ A, const _Float16* __restrict__ B,
    const float* __restrict__ bias, float* __restrict__ Cf,
    _Float16* __restrict__ Ch,
    int M, int N, int Kp, int lda, int ldb,
    int act, int acc, float cscale)
{
    const int wave = threadIdx.x >> 5;
    const int lane = threadIdx.x & 31;
    const int m0 = (blockIdx.y * 4 + wave) * 16;
    const int n0 = blockIdx.x * 32;
    if (m0 >= M) return;
    const int half = lane >> 4;
    const int mr = lane & 15;      // A row within tile
    const int nc = lane & 15;      // B/C column within tile

    // Per-lane fragment base pointers: 8 contiguous f16 per 128-bit load.
    const _Float16* Ap  = A + (size_t)(m0 + mr) * lda + half * 8;
    const _Float16* Bp0 = B + (size_t)(n0 + nc) * ldb + half * 8;
    const _Float16* Bp1 = B + (size_t)(n0 + 16 + nc) * ldb + half * 8;

    v8f c0 = {}, c1 = {};
    for (int k0 = 0; k0 < Kp; k0 += 32) {
        HFrag a, b0, b1;
        a.q[0]  = *(const v4u*)(Ap  + k0);
        a.q[1]  = *(const v4u*)(Ap  + k0 + 16);
        b0.q[0] = *(const v4u*)(Bp0 + k0);
        b0.q[1] = *(const v4u*)(Bp0 + k0 + 16);
        b1.q[0] = *(const v4u*)(Bp1 + k0);
        b1.q[1] = *(const v4u*)(Bp1 + k0 + 16);
        c0 = __builtin_amdgcn_wmma_f32_16x16x32_f16(
                false, a.v, false, b0.v, (short)0, c0, false, false);
        c1 = __builtin_amdgcn_wmma_f32_16x16x32_f16(
                false, a.v, false, b1.v, (short)0, c1, false, false);
    }

#pragma unroll
    for (int tile = 0; tile < 2; ++tile) {
        const v8f c = tile ? c1 : c0;
        const int n = n0 + tile * 16 + nc;
#pragma unroll
        for (int r = 0; r < 8; ++r) {
            const int m = m0 + r + half * 8;
            float v = c[r];
            if (bias) v += bias[n];
            if (act == ACT_RELU) {
                v = v > 0.f ? v : 0.f;
            } else if (act == ACT_SELU) {
                const float al = 1.6732632423543772f, sc = 1.0507009873554805f;
                v = sc * (v > 0.f ? v : al * (__expf(v) - 1.f));
            }
            if (Cf) {
                if (acc) Cf[(size_t)m * N + n] += cscale * v;
                else     Cf[(size_t)m * N + n] = v;
            }
            if (Ch) Ch[(size_t)m * N + n] = (_Float16)v;
        }
    }
}

// Zero-padded f32 -> f16 convert (no transpose): src (R x C) -> dst (Rp x Cp)
__global__ void cvtpad_kernel(const float* __restrict__ src, _Float16* __restrict__ dst,
                              int R, int C, int Rp, int Cp)
{
    long long i = (long long)blockIdx.x * blockDim.x + threadIdx.x;
    if (i >= (long long)Rp * Cp) return;
    int r = (int)(i / Cp), c = (int)(i % Cp);
    dst[i] = (_Float16)((r < R && c < C) ? src[(size_t)r * C + c] : 0.f);
}

// Transpose-convert: src (Kin x N) f32 -> dst (N x Kp) f16, K zero-padded.
__global__ void cvtpadT_kernel(const float* __restrict__ src, _Float16* __restrict__ dst,
                               int Kin, int N, int Kp)
{
    long long i = (long long)blockIdx.x * blockDim.x + threadIdx.x;
    if (i >= (long long)N * Kp) return;
    int n = (int)(i / Kp), k = (int)(i % Kp);
    dst[i] = (_Float16)((k < Kin) ? src[(size_t)k * N + n] : 0.f);
}

// Embedding gather: x16 (T x 352) = [word_table[wid], pos_table[pid], 0-pad]
__global__ void embed_kernel(const int* __restrict__ wid, const int* __restrict__ pid,
                             const float* __restrict__ wt, const float* __restrict__ pt,
                             _Float16* __restrict__ x16, int T)
{
    long long i = (long long)blockIdx.x * blockDim.x + threadIdx.x;
    if (i >= (long long)T * 352) return;
    int t = (int)(i / 352), c = (int)(i % 352);
    float v = 0.f;
    if (c < 300)      v = wt[(size_t)wid[t] * 300 + c];
    else if (c < 332) v = pt[(size_t)pid[t] * 32 + (c - 300)];
    x16[i] = (_Float16)v;
}

// Bi-LSTM scan: one workgroup, h in LDS, c in registers.
// Gx = x@Wih^T + bih (precomputed via WMMA); gates [i|f|g|o] of size 256 each.
__global__ __launch_bounds__(256) void lstm_scan_kernel(
    const float* __restrict__ Gx, const float* __restrict__ Whh,
    const float* __restrict__ bhh, float* __restrict__ hs, int T, int reverse)
{
    __shared__ float h[256];
    const int j = threadIdx.x;
    h[j] = 0.f;
    float cst = 0.f;
    __syncthreads();
    for (int s = 0; s < T; ++s) {
        const int t = reverse ? (T - 1 - s) : s;
        const float* gx = Gx + (size_t)t * 1024;
        float gi = gx[j]       + bhh[j];
        float gf = gx[256 + j] + bhh[256 + j];
        float gg = gx[512 + j] + bhh[512 + j];
        float go = gx[768 + j] + bhh[768 + j];
        const float* w0 = Whh + (size_t)j * 256;
        const float* w1 = Whh + (size_t)(256 + j) * 256;
        const float* w2 = Whh + (size_t)(512 + j) * 256;
        const float* w3 = Whh + (size_t)(768 + j) * 256;
        for (int k = 0; k < 256; ++k) {
            const float hv = h[k];
            gi += w0[k] * hv; gf += w1[k] * hv; gg += w2[k] * hv; go += w3[k] * hv;
        }
        const float si = 1.f / (1.f + __expf(-gi));
        const float sf = 1.f / (1.f + __expf(-gf));
        const float so = 1.f / (1.f + __expf(-go));
        cst = sf * cst + si * tanhf(gg);
        const float hn = so * tanhf(cst);
        __syncthreads();
        h[j] = hn;
        hs[(size_t)t * 256 + j] = hn;
        __syncthreads();
    }
}

// GRU scan, Hh=256. Gx = x@Wih^T + bih (T x 768), gates [r|z|n].
__global__ __launch_bounds__(256) void gru_scan_kernel(
    const float* __restrict__ Gx, const float* __restrict__ Whh,
    const float* __restrict__ bhh, const float* __restrict__ h0,
    float* __restrict__ hs, int T, int reverse)
{
    __shared__ float h[256];
    const int j = threadIdx.x;
    h[j] = h0 ? h0[j] : 0.f;
    __syncthreads();
    for (int s = 0; s < T; ++s) {
        const int t = reverse ? (T - 1 - s) : s;
        const float* gx = Gx + (size_t)t * 768;
        float hr = bhh[j], hz = bhh[256 + j], hn = bhh[512 + j];
        const float* w0 = Whh + (size_t)j * 256;
        const float* w1 = Whh + (size_t)(256 + j) * 256;
        const float* w2 = Whh + (size_t)(512 + j) * 256;
        for (int k = 0; k < 256; ++k) {
            const float hv = h[k];
            hr += w0[k] * hv; hz += w1[k] * hv; hn += w2[k] * hv;
        }
        const float r = 1.f / (1.f + __expf(-(gx[j] + hr)));
        const float z = 1.f / (1.f + __expf(-(gx[256 + j] + hz)));
        const float n = tanhf(gx[512 + j] + r * hn);
        const float hnew = (1.f - z) * n + z * h[j];
        __syncthreads();
        h[j] = hnew;
        hs[(size_t)t * 256 + j] = hnew;
        __syncthreads();
    }
}

// Decoder GRU scan, H=512, 2 units per thread. h0 = [ef[T-1] | eb[0]].
__global__ __launch_bounds__(256) void dec_scan_kernel(
    const float* __restrict__ Gx, const float* __restrict__ Whh,
    const float* __restrict__ bhh, const float* __restrict__ h0a,
    const float* __restrict__ h0b, float* __restrict__ dall, int T)
{
    __shared__ float h[512];
    const int j = threadIdx.x;
    h[j] = h0a[j];
    h[j + 256] = h0b[j];
    __syncthreads();
    for (int t = 0; t < T; ++t) {
        const float* gx = Gx + (size_t)t * 1536;
        float nv[2];
#pragma unroll
        for (int p = 0; p < 2; ++p) {
            const int u = j + p * 256;
            float hr = bhh[u], hz = bhh[512 + u], hn = bhh[1024 + u];
            const float* w0 = Whh + (size_t)u * 512;
            const float* w1 = Whh + (size_t)(512 + u) * 512;
            const float* w2 = Whh + (size_t)(1024 + u) * 512;
            for (int k = 0; k < 512; ++k) {
                const float hv = h[k];
                hr += w0[k] * hv; hz += w1[k] * hv; hn += w2[k] * hv;
            }
            const float r = 1.f / (1.f + __expf(-(gx[u] + hr)));
            const float z = 1.f / (1.f + __expf(-(gx[512 + u] + hz)));
            const float n = tanhf(gx[1024 + u] + r * hn);
            nv[p] = (1.f - z) * n + z * h[u];
        }
        __syncthreads();
        h[j] = nv[0];
        h[j + 256] = nv[1];
        dall[(size_t)t * 512 + j] = nv[0];
        dall[(size_t)t * 512 + j + 256] = nv[1];
        __syncthreads();
    }
}

// Concatenate fwd/bwd halves -> f32 and/or f16 copies (C = 2*Hh).
__global__ void concat_kernel(const float* __restrict__ a, const float* __restrict__ b,
                              float* __restrict__ o32, _Float16* __restrict__ o16,
                              int T, int Hh)
{
    long long i = (long long)blockIdx.x * blockDim.x + threadIdx.x;
    const int C = 2 * Hh;
    if (i >= (long long)T * C) return;
    int t = (int)(i / C), c = (int)(i % C);
    const float v = (c < Hh) ? a[(size_t)t * Hh + c] : b[(size_t)t * Hh + (c - Hh)];
    if (o32) o32[i] = v;
    if (o16) o16[i] = (_Float16)v;
}

// g_sum[t][l] = sum_t2 graph[t][t2][l]
__global__ void gsum_kernel(const float* __restrict__ graph, float* __restrict__ gsum,
                            int T, int L)
{
    long long i = (long long)blockIdx.x * blockDim.x + threadIdx.x;
    if (i >= (long long)T * L) return;
    int t = (int)(i / L), l = (int)(i % L);
    float s = 0.f;
    const float* g = graph + ((size_t)t * T) * L + l;
    for (int t2 = 0; t2 < T; ++t2) s += g[(size_t)t2 * L];
    gsum[i] = s;
}

__global__ void norm_kernel(const float* __restrict__ gsum, float* __restrict__ nrm,
                            int T, int L)
{
    int t = blockIdx.x * blockDim.x + threadIdx.x;
    if (t >= T) return;
    float s = 0.f;
    for (int l = 0; l < L; ++l) s += gsum[(size_t)t * L + l];
    nrm[t] = s + 1e-10f;
}

__global__ void gcn_mul_kernel(const float* __restrict__ tq, const float* __restrict__ rg,
                               _Float16* __restrict__ trg16, _Float16* __restrict__ rg16,
                               long long n)
{
    long long i = (long long)blockIdx.x * blockDim.x + threadIdx.x;
    if (i >= n) return;
    trg16[i] = (_Float16)(tq[i] * rg[i]);
    rg16[i]  = (_Float16)rg[i];
}

__global__ void gcn_update_kernel(const float* __restrict__ gc, const float* __restrict__ nrm,
                                  float* __restrict__ h32, _Float16* __restrict__ h16,
                                  long long n, int C)
{
    long long i = (long long)blockIdx.x * blockDim.x + threadIdx.x;
    if (i >= n) return;
    int t = (int)(i / C);
    float v = gc[i] / nrm[t];
    v = v > 0.f ? v : 0.f;
    h32[i] = v;
    h16[i] = (_Float16)v;
}

__global__ void gather_kernel(const float* __restrict__ dall, const int* __restrict__ idx,
                              _Float16* __restrict__ d16, int ND, int H)
{
    long long i = (long long)blockIdx.x * blockDim.x + threadIdx.x;
    if (i >= (long long)ND * H) return;
    int r = (int)(i / H), c = (int)(i % H);
    d16[i] = (_Float16)dall[(size_t)idx[r] * H + c];
}

__global__ void zero_kernel(float* __restrict__ p, long long n)
{
    long long i = (long long)blockIdx.x * blockDim.x + threadIdx.x;
    if (i < n) p[i] = 0.f;
}

// part2[e] = h_e[e,:] . u2
__global__ void p2_kernel(const _Float16* __restrict__ he, const float* __restrict__ u2,
                          float* __restrict__ p2, int T)
{
    int e = blockIdx.x * blockDim.x + threadIdx.x;
    if (e >= T) return;
    float s = 0.f;
    for (int l = 0; l < 128; ++l) s += (float)he[(size_t)e * 128 + l] * u2[l];
    p2[e] = s;
}

__global__ void addp2_kernel(float* __restrict__ acc, const float* __restrict__ p2,
                             const float* __restrict__ battn, int hh, int ND, int T)
{
    long long i = (long long)blockIdx.x * blockDim.x + threadIdx.x;
    if (i >= (long long)ND * T) return;
    acc[i] += 0.25f * (p2[i % T] + battn[hh]);
}

// Masked log-softmax per decode row (one block per row).
__global__ __launch_bounds__(256) void logsoftmax_kernel(
    const float* __restrict__ acc, const float* __restrict__ mask,
    float* __restrict__ out, int T)
{
    __shared__ float red[256];
    const int d = blockIdx.x, tid = threadIdx.x;
    float mx = -3.4e38f;
    for (int e = tid; e < T; e += 256) {
        const float m = mask[(size_t)d * T + e];
        const float v = acc[(size_t)d * T + e] * m + (1.f - m) * (-1e8f);
        mx = fmaxf(mx, v);
    }
    red[tid] = mx; __syncthreads();
    for (int s = 128; s > 0; s >>= 1) { if (tid < s) red[tid] = fmaxf(red[tid], red[tid + s]); __syncthreads(); }
    mx = red[0]; __syncthreads();
    float sum = 0.f;
    for (int e = tid; e < T; e += 256) {
        const float m = mask[(size_t)d * T + e];
        const float v = acc[(size_t)d * T + e] * m + (1.f - m) * (-1e8f);
        sum += __expf(v - mx);
    }
    red[tid] = sum; __syncthreads();
    for (int s = 128; s > 0; s >>= 1) { if (tid < s) red[tid] += red[tid + s]; __syncthreads(); }
    sum = red[0]; __syncthreads();
    const float ls = logf(sum);
    for (int e = tid; e < T; e += 256) {
        const float m = mask[(size_t)d * T + e];
        const float v = acc[(size_t)d * T + e] * m + (1.f - m) * (-1e8f);
        out[(size_t)d * T + e] = v - mx - ls;
    }
}

// ---------------------------------------------------------------------------

static inline dim3 g1(long long n, int b = 256) { return dim3((unsigned)((n + b - 1) / b)); }

static inline void gemm(hipStream_t st, const _Float16* A, const _Float16* B,
                        const float* bias, float* Cf, _Float16* Ch,
                        int M, int N, int Kp, int lda, int ldb,
                        int act, int acc, float cscale)
{
    dim3 grid(N / 32, (M + 63) / 64), block(128);
    wmma_gemm_kernel<<<grid, block, 0, st>>>(A, B, bias, Cf, Ch, M, N, Kp, lda, ldb,
                                             act, acc, cscale);
}

extern "C" void kernel_launch(void* const* d_in, const int* in_sizes, int n_in,
                              void* d_out, int out_size, void* d_ws, size_t ws_size,
                              hipStream_t stream)
{
    (void)in_sizes; (void)n_in; (void)out_size; (void)ws_size;
    const int T = 768, L = 45, H = 512, Hh = 256, ND = 64, XP = 352;

    const int*   wid   = (const int*)d_in[0];
    const int*   pid   = (const int*)d_in[1];
    const float* graph = (const float*)d_in[2];
    const int*   didx  = (const int*)d_in[3];
    const float* dmask = (const float*)d_in[4];
    const float* wtab  = (const float*)d_in[5];
    const float* ptab  = (const float*)d_in[6];
    float* out = (float*)d_out;

    char* W = (char*)d_ws; size_t off = 0;
    auto ALLOC = [&](size_t bytes) -> void* {
        void* p = W + off; off += (bytes + 255) & ~(size_t)255; return p;
    };
    _Float16* x16    = (_Float16*)ALLOC((size_t)T * XP * 2);
    _Float16* Wih16f = (_Float16*)ALLOC((size_t)1024 * XP * 2);   // N x Kp
    _Float16* Wih16b = (_Float16*)ALLOC((size_t)1024 * XP * 2);
    float*    Gxf    = (float*)ALLOC((size_t)T * 1024 * 4);
    float*    Gxb    = (float*)ALLOC((size_t)T * 1024 * 4);
    float*    hf     = (float*)ALLOC((size_t)T * Hh * 4);
    float*    hb     = (float*)ALLOC((size_t)T * Hh * 4);
    float*    h32    = (float*)ALLOC((size_t)T * H * 4);
    _Float16* h16    = (_Float16*)ALLOC((size_t)T * H * 2);
    float*    gsum   = (float*)ALLOC((size_t)T * L * 4);
    float*    gnrm   = (float*)ALLOC((size_t)T * 4);
    _Float16* gs16   = (_Float16*)ALLOC((size_t)T * 64 * 2);
    _Float16* rank16 = (_Float16*)ALLOC((size_t)64 * 64 * 2);     // N x Kp (T-converted)
    _Float16* Q16T   = (_Float16*)ALLOC((size_t)64 * 512 * 2);
    _Float16* P16T   = (_Float16*)ALLOC((size_t)512 * 64 * 2);
    _Float16* b16T   = (_Float16*)ALLOC((size_t)512 * 64 * 2);
    float*    rg32   = (float*)ALLOC((size_t)T * 64 * 4);
    float*    tq32   = (float*)ALLOC((size_t)T * 64 * 4);
    _Float16* rg16   = (_Float16*)ALLOC((size_t)T * 64 * 2);
    _Float16* trg16  = (_Float16*)ALLOC((size_t)T * 64 * 2);
    float*    gc32   = (float*)ALLOC((size_t)T * 512 * 4);
    _Float16* encW16f= (_Float16*)ALLOC((size_t)768 * 512 * 2);
    _Float16* encW16b= (_Float16*)ALLOC((size_t)768 * 512 * 2);
    float*    Gxef   = (float*)ALLOC((size_t)T * 768 * 4);
    float*    Gxeb   = (float*)ALLOC((size_t)T * 768 * 4);
    float*    ef     = (float*)ALLOC((size_t)T * Hh * 4);
    float*    eb     = (float*)ALLOC((size_t)T * Hh * 4);
    _Float16* e16    = (_Float16*)ALLOC((size_t)T * H * 2);
    _Float16* decW16 = (_Float16*)ALLOC((size_t)1536 * 512 * 2);
    float*    Gxd    = (float*)ALLOC((size_t)T * 1536 * 4);
    float*    dall   = (float*)ALLOC((size_t)T * H * 4);
    _Float16* d16    = (_Float16*)ALLOC((size_t)ND * H * 2);
    _Float16* W1hT   = (_Float16*)ALLOC((size_t)512 * 512 * 2);
    _Float16* W2hT   = (_Float16*)ALLOC((size_t)128 * 512 * 2);
    _Float16* Wd1hT  = (_Float16*)ALLOC((size_t)512 * 512 * 2);
    _Float16* Wd2hT  = (_Float16*)ALLOC((size_t)128 * 512 * 2);
    _Float16* u1hT   = (_Float16*)ALLOC((size_t)128 * 128 * 2);
    _Float16* he1    = (_Float16*)ALLOC((size_t)T * 512 * 2);
    _Float16* he     = (_Float16*)ALLOC((size_t)T * 128 * 2);
    _Float16* hd1    = (_Float16*)ALLOC((size_t)ND * 512 * 2);
    _Float16* hd     = (_Float16*)ALLOC((size_t)ND * 128 * 2);
    _Float16* tmp16  = (_Float16*)ALLOC((size_t)T * 128 * 2);
    float*    p2     = (float*)ALLOC((size_t)T * 4);
    float*    acc    = (float*)ALLOC((size_t)ND * T * 4);

    // --- Embedding + LSTM input projections (WMMA) ---
    embed_kernel<<<g1((long long)T * XP), 256, 0, stream>>>(wid, pid, wtab, ptab, x16, T);
    // lstm_Wih_* are (1024 x 332) = (N x K): pad-convert straight to N x Kp.
    cvtpad_kernel<<<g1((long long)1024 * XP), 256, 0, stream>>>((const float*)d_in[7],  Wih16f, 1024, 332, 1024, XP);
    cvtpad_kernel<<<g1((long long)1024 * XP), 256, 0, stream>>>((const float*)d_in[11], Wih16b, 1024, 332, 1024, XP);
    gemm(stream, x16, Wih16f, (const float*)d_in[9],  Gxf, nullptr, T, 1024, XP, XP, XP, ACT_NONE, 0, 1.f);
    gemm(stream, x16, Wih16b, (const float*)d_in[13], Gxb, nullptr, T, 1024, XP, XP, XP, ACT_NONE, 0, 1.f);
    lstm_scan_kernel<<<1, 256, 0, stream>>>(Gxf, (const float*)d_in[8],  (const float*)d_in[10], hf, T, 0);
    lstm_scan_kernel<<<1, 256, 0, stream>>>(Gxb, (const float*)d_in[12], (const float*)d_in[14], hb, T, 1);
    concat_kernel<<<g1((long long)T * H), 256, 0, stream>>>(hf, hb, h32, h16, T, Hh);

    // --- GCN ---
    gsum_kernel<<<g1((long long)T * L), 256, 0, stream>>>(graph, gsum, T, L);
    norm_kernel<<<g1(T), 256, 0, stream>>>(gsum, gnrm, T, L);
    cvtpad_kernel<<<g1((long long)T * 64), 256, 0, stream>>>(gsum, gs16, T, L, T, 64);
    for (int l = 0; l < 2; ++l) {
        const float* rk = (const float*)d_in[15] + (size_t)l * 45 * 64;   // (K=45, N=64)
        const float* Qw = (const float*)d_in[16] + (size_t)l * 512 * 64;  // (K=512, N=64)
        const float* Pw = (const float*)d_in[17] + (size_t)l * 64 * 512;  // (K=64, N=512)
        const float* bw = (const float*)d_in[18] + (size_t)l * 64 * 512;
        cvtpadT_kernel<<<g1(64 * 64), 256, 0, stream>>>(rk, rank16, 45, 64, 64);
        cvtpadT_kernel<<<g1(64 * 512), 256, 0, stream>>>(Qw, Q16T, 512, 64, 512);
        cvtpadT_kernel<<<g1(512 * 64), 256, 0, stream>>>(Pw, P16T, 64, 512, 64);
        cvtpadT_kernel<<<g1(512 * 64), 256, 0, stream>>>(bw, b16T, 64, 512, 64);
        gemm(stream, gs16, rank16, nullptr, rg32, nullptr, T, 64, 64, 64, 64, ACT_NONE, 0, 1.f);
        gemm(stream, h16, Q16T, nullptr, tq32, nullptr, T, 64, 512, 512, 512, ACT_NONE, 0, 1.f);
        gcn_mul_kernel<<<g1((long long)T * 64), 256, 0, stream>>>(tq32, rg32, trg16, rg16, (long long)T * 64);
        gemm(stream, trg16, P16T, nullptr, gc32, nullptr, T, 512, 64, 64, 64, ACT_NONE, 0, 1.f);
        gemm(stream, rg16, b16T, nullptr, gc32, nullptr, T, 512, 64, 64, 64, ACT_NONE, 1, 1.f);
        gcn_update_kernel<<<g1((long long)T * 512), 256, 0, stream>>>(gc32, gnrm, h32, h16, (long long)T * 512, 512);
    }

    // --- Bi-GRU encoder ---
    // enc_Wih_* are (768 x 512) = (N x K): straight convert.
    cvtpad_kernel<<<g1((long long)768 * 512), 256, 0, stream>>>((const float*)d_in[19], encW16f, 768, 512, 768, 512);
    cvtpad_kernel<<<g1((long long)768 * 512), 256, 0, stream>>>((const float*)d_in[23], encW16b, 768, 512, 768, 512);
    gemm(stream, h16, encW16f, (const float*)d_in[21], Gxef, nullptr, T, 768, 512, 512, 512, ACT_NONE, 0, 1.f);
    gemm(stream, h16, encW16b, (const float*)d_in[25], Gxeb, nullptr, T, 768, 512, 512, 512, ACT_NONE, 0, 1.f);
    gru_scan_kernel<<<1, 256, 0, stream>>>(Gxef, (const float*)d_in[20], (const float*)d_in[22], nullptr, ef, T, 0);
    gru_scan_kernel<<<1, 256, 0, stream>>>(Gxeb, (const float*)d_in[24], (const float*)d_in[26], nullptr, eb, T, 1);
    concat_kernel<<<g1((long long)T * H), 256, 0, stream>>>(ef, eb, nullptr, e16, T, Hh);

    // --- Decoder GRU (h0 = [ef[T-1] | eb[0]]) ---
    cvtpad_kernel<<<g1((long long)1536 * 512), 256, 0, stream>>>((const float*)d_in[27], decW16, 1536, 512, 1536, 512);
    gemm(stream, h16, decW16, (const float*)d_in[29], Gxd, nullptr, T, 1536, 512, 512, 512, ACT_NONE, 0, 1.f);
    dec_scan_kernel<<<1, 256, 0, stream>>>(Gxd, (const float*)d_in[28], (const float*)d_in[30],
                                           ef + (size_t)(T - 1) * Hh, eb, dall, T);
    gather_kernel<<<g1((long long)ND * H), 256, 0, stream>>>(dall, didx, d16, ND, H);

    // --- Bi-affine attention heads ---
    zero_kernel<<<g1((long long)ND * T), 256, 0, stream>>>(acc, (long long)ND * T);
    for (int hh = 0; hh < 4; ++hh) {
        const float* We1 = (const float*)d_in[31] + (size_t)hh * 512 * 512; // (K,N)
        const float* We2 = (const float*)d_in[32] + (size_t)hh * 512 * 128;
        const float* Wd1 = (const float*)d_in[33] + (size_t)hh * 512 * 512;
        const float* Wd2 = (const float*)d_in[34] + (size_t)hh * 512 * 128;
        const float* u1  = (const float*)d_in[35] + (size_t)hh * 128 * 128;
        const float* u2  = (const float*)d_in[36] + (size_t)hh * 128;
        cvtpadT_kernel<<<g1(512 * 512), 256, 0, stream>>>(We1, W1hT, 512, 512, 512);
        cvtpadT_kernel<<<g1(128 * 512), 256, 0, stream>>>(We2, W2hT, 512, 128, 512);
        cvtpadT_kernel<<<g1(512 * 512), 256, 0, stream>>>(Wd1, Wd1hT, 512, 512, 512);
        cvtpadT_kernel<<<g1(128 * 512), 256, 0, stream>>>(Wd2, Wd2hT, 512, 128, 512);
        cvtpadT_kernel<<<g1(128 * 128), 256, 0, stream>>>(u1, u1hT, 128, 128, 128);
        gemm(stream, e16, W1hT, nullptr, nullptr, he1, T, 512, 512, 512, 512, ACT_SELU, 0, 1.f);
        gemm(stream, he1, W2hT, nullptr, nullptr, he, T, 128, 512, 512, 512, ACT_SELU, 0, 1.f);
        gemm(stream, d16, Wd1hT, nullptr, nullptr, hd1, ND, 512, 512, 512, 512, ACT_SELU, 0, 1.f);
        gemm(stream, hd1, Wd2hT, nullptr, nullptr, hd, ND, 128, 512, 512, 512, ACT_SELU, 0, 1.f);
        gemm(stream, he, u1hT, nullptr, nullptr, tmp16, T, 128, 128, 128, 128, ACT_NONE, 0, 1.f);
        // part1: acc += 0.25 * h_d @ tmp^T   (tmp already stored N x K)
        gemm(stream, hd, tmp16, nullptr, acc, nullptr, ND, T, 128, 128, 128, ACT_NONE, 1, 0.25f);
        p2_kernel<<<g1(T), 256, 0, stream>>>(he, u2, p2, T);
        addp2_kernel<<<g1((long long)ND * T), 256, 0, stream>>>(acc, p2, (const float*)d_in[38], hh, ND, T);
    }

    // --- Masked log-softmax over e (axis=2) ---
    logsoftmax_kernel<<<ND, 256, 0, stream>>>(acc, dmask, out, T);
}
